// LatentSDE_50079318671668
// MI455X (gfx1250) — compile-verified, matrix operand order checked by reading, not run on hardware
//
#include <hip/hip_runtime.h>
#include <hip/hip_bf16.h>

// ---------------------------------------------------------------------------
// LatentSDE on MI455X (gfx1250): per-workgroup 16-row batch slice runs the
// whole encoder + T=32 scan. GEMMs via v_wmma_f32_16x16x32_bf16 (wave32).
// Small weights staged into LDS with global_load_async_to_lds_b128 (CDNA5),
// overlapped with the H x H WMMA GEMMs; big W2 matrices stream from L2.
// ---------------------------------------------------------------------------

typedef __attribute__((ext_vector_type(16))) __bf16 v16bf;
typedef __attribute__((ext_vector_type(8)))  float  v8f;
typedef __attribute__((__vector_size__(16))) int    v4i;

#define TT 32
#define BB 1024
#define DD 32
#define LL 32
#define HH 256
#define AA 8
#define H_STEP 0.015625f          // (T1-T0)/T = 0.5/32
#define SQRT_H 0.125f
#define EPSV   1e-7f
#define LP_C0  3.6862316527834188f  // -log(0.01) - 0.5*log(2*pi)

// ---- bf16 workspace offsets (in bf16 elements), all weights stored [N][K] --
#define OFF_E1  0u        // enc_W1t [256][32]
#define OFF_E2  8192u     // enc_W2t [256][256]
#define OFF_E3  73728u    // enc_W3t [32][256]
#define OFF_QZ  81920u    // qz_Wt   [64][32]
#define OFF_F1  83968u
#define OFF_F2  92160u
#define OFF_F3  157696u
#define OFF_H1  165888u
#define OFF_H2  174080u
#define OFF_H3  239616u
#define OFF_P1  247808u
#define OFF_P2  256000u
#define OFF_P3  321536u
#define OFF_AE  329728u   // ae_Wt [32][96] (K padded 72->96 with zeros)
#define OFF_G1  332800u   // g_W1t [256][32] (i.e. [h][l])
#define OFF_GB1 340992u
#define OFF_G2  349184u

struct SDEArgs {
  const float* xs; const float* actions; const float* xs_target;
  const float* z0n; const float* dWn;
  const float *enc_b1,*enc_b2,*enc_b3,*qz_b;
  const float *f_b1,*f_b2,*f_b3,*h_b1,*h_b2,*h_b3;
  const float *g_b2,*p_b1,*p_b2,*p_b3,*ae_b;
  const __bf16 *eW1,*eW2,*eW3,*qzW,*fW1,*fW2,*fW3,*hW1,*hW2,*hW3;
  const __bf16 *pW1,*pW2,*pW3,*aeW,*gW1,*gB1,*gW2;
  float* out;
};

// ---- CDNA5 async global->LDS staging (guarded; sync fallback otherwise) ----
#if __has_builtin(__builtin_amdgcn_global_load_async_to_lds_b128)
#define HAVE_ASYNC_LDS 1
#else
#define HAVE_ASYNC_LDS 0
#endif

typedef __attribute__((address_space(1))) v4i* gv4i_p;
typedef __attribute__((address_space(3))) v4i* lv4i_p;

__device__ __forceinline__ void stage_weight(const __bf16* __restrict__ g,
                                             __bf16* __restrict__ l, int bytes) {
  const int tid = threadIdx.x;
#if HAVE_ASYNC_LDS
  for (int off = tid * 16; off < bytes; off += 256 * 16) {
    __builtin_amdgcn_global_load_async_to_lds_b128(
        (gv4i_p)((const char*)g + off),
        (lv4i_p)((char*)l + off), 0, 0);
  }
#else
  for (int off = tid * 16; off < bytes; off += 256 * 16)
    *(uint4*)((char*)l + off) = *(const uint4*)((const char*)g + off);
#endif
}

__device__ __forceinline__ void stage_wait() {
#if HAVE_ASYNC_LDS
#if __has_builtin(__builtin_amdgcn_s_wait_asynccnt)
  __builtin_amdgcn_s_wait_asynccnt(0);
#else
  asm volatile("s_wait_asynccnt 0" ::: "memory");
#endif
#endif
  __syncthreads();
}

// -------- fragment load: two contiguous 16B chunks per lane -----------------
__device__ __forceinline__ v16bf load_frag(const __bf16* __restrict__ p,
                                           int ldk, int row0, int k0, int lane) {
  int r  = row0 + (lane & 15);
  int kh = (lane >> 4) << 3;
  const __bf16* b0 = p + r * ldk + k0 + kh;
  v16bf o;
#pragma unroll
  for (int j = 0; j < 8; ++j) { o[j] = b0[j]; o[j + 8] = b0[j + 16]; }
  return o;
}

#define ACT_NONE 0
#define ACT_RELU 1
#define ACT_TANH 2

// ---- workgroup GEMM: [16 x K](LDS bf16) @ Wt[N][K](bf16, LDS or global) ----
template <int ACT, bool WB, bool WF, int K>
__device__ __forceinline__ void wg_gemm(const __bf16* __restrict__ X,
                                        const __bf16* __restrict__ Wt,
                                        const float* __restrict__ bias, int N,
                                        __bf16* __restrict__ Yb,
                                        float* __restrict__ Yf) {
  const int tid = threadIdx.x, wid = tid >> 5, lane = tid & 31;
  const int ntiles = N >> 4;
  for (int nt = wid; nt < ntiles; nt += 8) {
    v8f acc;
#pragma unroll
    for (int j = 0; j < 8; ++j) acc[j] = 0.f;
#pragma unroll
    for (int k0 = 0; k0 < K; k0 += 32) {
      v16bf a = load_frag(X, K, 0, k0, lane);
      v16bf b = load_frag(Wt, K, nt << 4, k0, lane);
      acc = __builtin_amdgcn_wmma_f32_16x16x32_bf16(false, a, false, b,
                                                    (short)0, acc, false, false);
    }
    const int n = (nt << 4) + (lane & 15);
    const float bv = bias[n];
    const int mb = (lane >> 4) << 3;
#pragma unroll
    for (int v = 0; v < 8; ++v) {
      float x = acc[v] + bv;
      if (ACT == ACT_RELU) x = fmaxf(x, 0.f);
      if (ACT == ACT_TANH) x = tanhf(x);
      const int m = mb + v;
      if (WB) Yb[m * N + n] = (__bf16)x;
      if (WF) Yf[m * N + n] = x;
    }
  }
}

// -------- last proj layer: stream to global + log-likelihood accumulation --
__device__ __forceinline__ void wg_gemm_proj_out(const __bf16* __restrict__ X,
                                                 const __bf16* __restrict__ Wt,
                                                 const float* __restrict__ bias,
                                                 float* __restrict__ outG,
                                                 const float* __restrict__ tgt,
                                                 float& lpAcc) {
  const int tid = threadIdx.x, wid = tid >> 5, lane = tid & 31;
  if (wid < 2) {
    v8f acc;
#pragma unroll
    for (int j = 0; j < 8; ++j) acc[j] = 0.f;
#pragma unroll
    for (int k0 = 0; k0 < HH; k0 += 32) {
      v16bf a = load_frag(X, HH, 0, k0, lane);
      v16bf b = load_frag(Wt, HH, wid << 4, k0, lane);
      acc = __builtin_amdgcn_wmma_f32_16x16x32_bf16(false, a, false, b,
                                                    (short)0, acc, false, false);
    }
    const int n = (wid << 4) + (lane & 15);
    const float bv = bias[n];
    const int mb = (lane >> 4) << 3;
#pragma unroll
    for (int v = 0; v < 8; ++v) {
      const int m = mb + v;
      const float x = acc[v] + bv;
      const int idx = m * DD + n;
      outG[idx] = x;
      const float d = (tgt[idx] - x) * 100.f;   // / NOISE_STD
      lpAcc += -0.5f * d * d + LP_C0;
    }
  }
}

__device__ __forceinline__ float safe_g(float g) {
  return (fabsf(g) > EPSV) ? g : (g >= 0.f ? EPSV : -EPSV);
}

// MLP3 with async W1/W3 staging: W3's async copy overlaps the W2 WMMA GEMM.
template <int A1>
__device__ __forceinline__ void mlp3(const __bf16* in32, const __bf16* W1g,
                                     const float* b1, const __bf16* W2,
                                     const float* b2, const __bf16* W3g,
                                     const float* b3, __bf16* sX, __bf16* sY,
                                     __bf16* sW, float* out32) {
  stage_weight(W1g, sW, HH * LL * 2);
  stage_wait();
  wg_gemm<A1, true, false, LL>(in32, sW, b1, HH, sX, nullptr);
  __syncthreads();
  stage_weight(W3g, sW, HH * LL * 2);                 // overlaps W2 GEMM
  wg_gemm<ACT_TANH, true, false, HH>(sX, W2, b2, HH, sY, nullptr);
  stage_wait();
  wg_gemm<ACT_NONE, false, true, HH>(sY, sW, b3, LL, nullptr, out32);
  __syncthreads();
}

__global__ __launch_bounds__(256) void sde_main(SDEArgs P) {
  __shared__ __align__(16) __bf16 sX[16 * HH];
  __shared__ __align__(16) __bf16 sY[16 * HH];
  __shared__ __align__(16) __bf16 sIn[16 * 96];
  __shared__ __align__(16) __bf16 sZ[16 * LL];
  __shared__ __align__(16) __bf16 sW[HH * LL];        // weight staging (16KB)
  __shared__ float sPool[512 * 10];
  __shared__ float sLP, sLR;

  float* pYst = sPool + 0 * 512;
  float* pY0  = sPool + 1 * 512;
  float* pZh  = sPool + 2 * 512;
  float* pDw  = sPool + 3 * 512;
  float* pFa  = sPool + 4 * 512;
  float* pHa  = sPool + 5 * 512;
  float* pGa  = sPool + 6 * 512;
  float* pFb  = sPool + 7 * 512;
  float* pHb  = sPool + 8 * 512;
  float* pGb  = sPool + 9 * 512;
  float* pQz  = pFa;                                  // encoder-only alias

  const int tid = threadIdx.x;
  const int r0 = blockIdx.x * 16;
  float accLP = 0.f, accLR = 0.f;

  if (tid == 0) { sLP = 0.f; sLR = 0.f; }

  // warm L2/WGP$ for the big weights (global_prefetch_b8)
  {
    const char* ws[3] = {(const char*)P.fW2, (const char*)P.hW2, (const char*)P.pW2};
#pragma unroll
    for (int w = 0; w < 3; ++w)
      for (size_t off = (size_t)tid * 64; off < (size_t)HH * HH * 2; off += 256 * 64)
        __builtin_prefetch(ws[w] + off, 0, 1);
  }

  // ---------------- encoder on xs[0] rows [r0, r0+16) ----------------------
  stage_weight(P.eW1, sW, HH * DD * 2);
  for (int i = tid; i < 16 * DD; i += 256) {
    const int r = i >> 5, d = i & 31;
    sZ[i] = (__bf16)P.xs[(r0 + r) * DD + d];
  }
  stage_wait();
  wg_gemm<ACT_RELU, true, false, DD>(sZ, sW, P.enc_b1, HH, sX, nullptr);
  __syncthreads();
  stage_weight(P.eW3, sW, HH * DD * 2);
  wg_gemm<ACT_TANH, true, false, HH>(sX, P.eW2, P.enc_b2, HH, sY, nullptr);
  stage_wait();
  wg_gemm<ACT_NONE, true, false, HH>(sY, sW, P.enc_b3, DD, sZ, nullptr);
  __syncthreads();
  stage_weight(P.qzW, sW, 64 * DD * 2);
  stage_wait();
  wg_gemm<ACT_NONE, false, true, DD>(sZ, sW, P.qz_b, 64, nullptr, pQz);
  __syncthreads();

  for (int i = tid; i < 512; i += 256) {
    const int r = i >> 5, l = i & 31;
    const float mean = pQz[r * 64 + l];
    const float lst  = pQz[r * 64 + 32 + l];
    pYst[i] = mean + expf(lst) * P.z0n[(r0 + r) * LL + l];
    accLR += -lst + 0.5f * (expf(2.f * lst) + mean * mean) - 0.5f;  // KL
  }
  __syncthreads();

  // ---------------- scan over T steps --------------------------------------
#pragma unroll 1
  for (int t = 0; t < TT; ++t) {
    const int ta = (t == 0) ? 0 : (t - 1);
    stage_weight(P.aeW, sW, LL * 96 * 2);
    // concat [z_prev | a | x | pad] -> bf16 [16][96]
    for (int i = tid; i < 16 * 96; i += 256) {
      const int r = i / 96, c = i % 96;
      float v;
      if (c < 32)      v = pYst[r * 32 + c];
      else if (c < 40) v = P.actions[(size_t)ta * BB * AA + (r0 + r) * AA + (c - 32)];
      else if (c < 72) v = P.xs[(size_t)ta * BB * DD + (r0 + r) * DD + (c - 40)];
      else             v = 0.f;
      sIn[i] = (__bf16)v;
    }
    stage_wait();
    wg_gemm<ACT_NONE, true, true, 96>(sIn, sW, P.ae_b, LL, sZ, pY0);  // y0
    __syncthreads();

    mlp3<ACT_RELU>(sZ, P.fW1, P.f_b1, P.fW2, P.f_b2, P.fW3, P.f_b3, sX, sY, sW, pFa);
    mlp3<ACT_RELU>(sZ, P.hW1, P.h_b1, P.hW2, P.h_b2, P.hW3, P.h_b3, sX, sY, sW, pHa);

    // g(y0): per (row, latent) H-wide dot, coalesced via transposed g weights
    for (int i = tid; i < 512; i += 256) {
      const int l = i & 31;
      const float y = pY0[i];
      float acc = 0.f;
      for (int hh = 0; hh < HH; ++hh) {
        const float hv = fmaxf(y * (float)P.gW1[hh * 32 + l] +
                               (float)P.gB1[hh * 32 + l], 0.f);
        acc += hv * (float)P.gW2[hh * 32 + l];
      }
      pGa[i] = acc + P.g_b2[l];
    }
    __syncthreads();

    // pass 1: u0, zh
    for (int i = tid; i < 512; i += 256) {
      const int r = i >> 5;
      const float dw = SQRT_H * P.dWn[(size_t)t * BB * LL + (r0 + r) * LL + (i & 31)];
      pDw[i] = dw;
      const float u = (pFa[i] - pHa[i]) / safe_g(pGa[i]);
      accLR += 0.25f * H_STEP * u * u;
      const float zh = pY0[i] + H_STEP * pFa[i] + pGa[i] * dw;
      pZh[i] = zh;
      sZ[i] = (__bf16)zh;
    }
    __syncthreads();

    mlp3<ACT_RELU>(sZ, P.fW1, P.f_b1, P.fW2, P.f_b2, P.fW3, P.f_b3, sX, sY, sW, pFb);
    mlp3<ACT_RELU>(sZ, P.hW1, P.h_b1, P.hW2, P.h_b2, P.hW3, P.h_b3, sX, sY, sW, pHb);

    for (int i = tid; i < 512; i += 256) {
      const int l = i & 31;
      const float y = pZh[i];
      float acc = 0.f;
      for (int hh = 0; hh < HH; ++hh) {
        const float hv = fmaxf(y * (float)P.gW1[hh * 32 + l] +
                               (float)P.gB1[hh * 32 + l], 0.f);
        acc += hv * (float)P.gW2[hh * 32 + l];
      }
      pGb[i] = acc + P.g_b2[l];
    }
    __syncthreads();

    // pass 2: u1, y1
    for (int i = tid; i < 512; i += 256) {
      const float u = (pFb[i] - pHb[i]) / safe_g(pGb[i]);
      accLR += 0.25f * H_STEP * u * u;
      const float y1 = pY0[i] + 0.5f * H_STEP * (pFa[i] + pFb[i]) +
                       0.5f * (pGa[i] + pGb[i]) * pDw[i];
      pYst[i] = y1;
      sZ[i] = (__bf16)y1;
    }
    __syncthreads();

    // proj MLP, last layer streams predicted_xs + accumulates log p(x)
    stage_weight(P.pW1, sW, HH * LL * 2);
    stage_wait();
    wg_gemm<ACT_TANH, true, false, LL>(sZ, sW, P.p_b1, HH, sX, nullptr);
    __syncthreads();
    stage_weight(P.pW3, sW, HH * LL * 2);               // overlaps W2 GEMM
    wg_gemm<ACT_TANH, true, false, HH>(sX, P.pW2, P.p_b2, HH, sY, nullptr);
    stage_wait();
    wg_gemm_proj_out(sY, sW, P.p_b3,
                     P.out + 2 + (size_t)t * BB * DD + (size_t)r0 * DD,
                     P.xs_target + (size_t)t * BB * DD + (size_t)r0 * DD, accLP);
    __syncthreads();
  }

  atomicAdd(&sLP, accLP);
  atomicAdd(&sLR, accLR);
  __syncthreads();
  if (tid == 0) {
    atomicAdd(P.out + 0, sLP * (1.f / (float)BB));
    atomicAdd(P.out + 1, sLR * (1.f / (float)BB));
  }
}

// ---- weight conversion: fp32 [K][N] -> bf16 transposed [N][Kpad] -----------
__global__ void convert_t(const float* __restrict__ src, __bf16* __restrict__ dst,
                          int K, int N, int Kpad) {
  const int i = blockIdx.x * blockDim.x + threadIdx.x;
  if (i >= N * Kpad) return;
  const int n = i / Kpad, k = i % Kpad;
  dst[i] = (k < K) ? (__bf16)src[k * N + n] : (__bf16)0.f;
}

__global__ void zero_scalars(float* o) {
  if (threadIdx.x == 0) { o[0] = 0.f; o[1] = 0.f; }
}

extern "C" void kernel_launch(void* const* d_in, const int* in_sizes, int n_in,
                              void* d_out, int out_size, void* d_ws, size_t ws_size,
                              hipStream_t stream) {
  (void)in_sizes; (void)n_in; (void)out_size; (void)ws_size;
  __bf16* W = (__bf16*)d_ws;

  auto cv = [&](int pi, unsigned off, int K, int N, int Kpad) {
    const int total = N * Kpad;
    convert_t<<<(total + 255) / 256, 256, 0, stream>>>(
        (const float*)d_in[pi], W + off, K, N, Kpad);
  };

  zero_scalars<<<1, 64, 0, stream>>>((float*)d_out);

  cv(3,  OFF_E1, 32, 256, 32);   // enc_W1
  cv(5,  OFF_E2, 256, 256, 256); // enc_W2
  cv(7,  OFF_E3, 256, 32, 256);  // enc_W3
  cv(9,  OFF_QZ, 32, 64, 32);    // qz_W
  cv(11, OFF_F1, 32, 256, 32);   // f_W1
  cv(13, OFF_F2, 256, 256, 256); // f_W2
  cv(15, OFF_F3, 256, 32, 256);  // f_W3
  cv(17, OFF_H1, 32, 256, 32);   // h_W1
  cv(19, OFF_H2, 256, 256, 256); // h_W2
  cv(21, OFF_H3, 256, 32, 256);  // h_W3
  cv(27, OFF_P1, 32, 256, 32);   // proj_W1
  cv(29, OFF_P2, 256, 256, 256); // proj_W2
  cv(31, OFF_P3, 256, 32, 256);  // proj_W3
  cv(33, OFF_AE, 72, 32, 96);    // ae_W (K padded to 96)
  cv(23, OFF_G1, 32, 256, 32);   // g_W1 -> [h][l]
  cv(24, OFF_GB1, 32, 256, 32);  // g_b1 -> [h][l]
  cv(25, OFF_G2, 32, 256, 32);   // g_W2 -> [h][l]

  SDEArgs P;
  P.xs = (const float*)d_in[0];
  P.actions = (const float*)d_in[1];
  P.xs_target = (const float*)d_in[2];
  P.z0n = (const float*)d_in[37];
  P.dWn = (const float*)d_in[38];
  P.enc_b1 = (const float*)d_in[4];  P.enc_b2 = (const float*)d_in[6];
  P.enc_b3 = (const float*)d_in[8];  P.qz_b   = (const float*)d_in[10];
  P.f_b1 = (const float*)d_in[12];   P.f_b2 = (const float*)d_in[14];
  P.f_b3 = (const float*)d_in[16];
  P.h_b1 = (const float*)d_in[18];   P.h_b2 = (const float*)d_in[20];
  P.h_b3 = (const float*)d_in[22];
  P.g_b2 = (const float*)d_in[26];
  P.p_b1 = (const float*)d_in[28];   P.p_b2 = (const float*)d_in[30];
  P.p_b3 = (const float*)d_in[32];
  P.ae_b = (const float*)d_in[34];
  P.eW1 = W + OFF_E1; P.eW2 = W + OFF_E2; P.eW3 = W + OFF_E3;
  P.qzW = W + OFF_QZ;
  P.fW1 = W + OFF_F1; P.fW2 = W + OFF_F2; P.fW3 = W + OFF_F3;
  P.hW1 = W + OFF_H1; P.hW2 = W + OFF_H2; P.hW3 = W + OFF_H3;
  P.pW1 = W + OFF_P1; P.pW2 = W + OFF_P2; P.pW3 = W + OFF_P3;
  P.aeW = W + OFF_AE;
  P.gW1 = W + OFF_G1; P.gB1 = W + OFF_GB1; P.gW2 = W + OFF_G2;
  P.out = (float*)d_out;

  sde_main<<<BB / 16, 256, 0, stream>>>(P);
}
